// Pct_78271484002682
// MI455X (gfx1250) — matrix operand order repaired and checked by reading, not compiled, not run on hardware
//
#include <hip/hip_runtime.h>
#include <hip/hip_bf16.h>

// ---------------------------------------------------------------------------
// Types for CDNA5 WMMA (wave32): v_wmma_f32_16x16x32_f16
// ---------------------------------------------------------------------------
typedef _Float16 v16h __attribute__((ext_vector_type(16)));
typedef _Float16 h8   __attribute__((ext_vector_type(8)));
typedef float    v8f  __attribute__((ext_vector_type(8)));
typedef int      v4i  __attribute__((ext_vector_type(4)));

union H16 { v16h v; h8 h[2]; };

__device__ __forceinline__ v8f wmma_f16(v16h a, v16h b, v8f c) {
  // 8 args: (neg_a, A, neg_b, B, c_mod, C, reuse_a, reuse_b)
  return __builtin_amdgcn_wmma_f32_16x16x32_f16(false, a, false, b, (short)0, c,
                                                false, false);
}
__device__ __forceinline__ h8 h8zero() {
  h8 z;
#pragma unroll
  for (int i = 0; i < 8; ++i) z[i] = (_Float16)0.f;
  return z;
}
__device__ __forceinline__ v8f v8fzero() {
  v8f z;
#pragma unroll
  for (int i = 0; i < 8; ++i) z[i] = 0.f;
  return z;
}

// CDNA5 async global->LDS copy (ASYNCcnt-tracked). Guarded: fall back to
// direct global loads + global_prefetch_b8 if the builtin is unavailable.
#if __has_builtin(__builtin_amdgcn_global_load_async_to_lds_b128) && \
    __has_builtin(__builtin_amdgcn_s_wait_asynccnt)
#define USE_ASYNC_LDS 1
__device__ __forceinline__ void async_b128(const void* g, void* l) {
  typedef __attribute__((address_space(1))) v4i* gp_t;   // global int4*
  typedef __attribute__((address_space(3))) v4i* lp_t;   // LDS int4*
  __builtin_amdgcn_global_load_async_to_lds_b128((gp_t)g, (lp_t)l, 0, 0);
}
#else
#define USE_ASYNC_LDS 0
#endif

// ---------------------------------------------------------------------------
// Problem constants
// ---------------------------------------------------------------------------
constexpr int cB = 8, cC = 256, cN = 1024, cD = 256;
constexpr int cP = 1025;          // tokens (cls + 1024)
constexpr int cPP = 1056;         // padded to multiple of 32
constexpr int cH = 8, cDK = 16, cDV = 32;
constexpr int cHDK = cH * cDK;    // 128
constexpr int cHDV = cH * cDV;    // 256
constexpr int cNB = 4;

// ---------------------------------------------------------------------------
// f32 -> f16 convert
// ---------------------------------------------------------------------------
__global__ void k_cvt(const float* __restrict__ s, _Float16* __restrict__ d, int n) {
  int i = blockIdx.x * 256 + threadIdx.x;
  if (i < n) d[i] = (_Float16)s[i];
}

// x [B,C,N] f32 -> xt [B,N,C] f16
__global__ void k_transpose_x(const float* __restrict__ x, _Float16* __restrict__ xt) {
  int idx = blockIdx.x * 256 + threadIdx.x;          // B*N*C = 2M
  if (idx >= cB * cN * cC) return;
  int c = idx & 255;
  int rest = idx >> 8;
  int n = rest & 1023;
  int b = rest >> 10;
  xt[idx] = (_Float16)x[((size_t)b * cC + c) * cN + n];
}

// h[b][0][:] = cls_token
__global__ void k_cls(const float* __restrict__ cls, float* __restrict__ h) {
  int i = blockIdx.x * 256 + threadIdx.x;            // B*D
  if (i < cB * cD) {
    int b = i >> 8, d = i & 255;
    h[(size_t)b * cPP * cD + d] = cls[d];
  }
}

// h [B,PP,D] -> out [B,P,D]
__global__ void k_out(const float* __restrict__ h, float* __restrict__ out) {
  size_t i = (size_t)blockIdx.x * 256 + threadIdx.x; // B*P*D
  if (i >= (size_t)cB * cP * cD) return;
  size_t d = i & 255;
  size_t rest = i >> 8;
  size_t p = rest % cP;
  size_t b = rest / cP;
  out[i] = h[((size_t)b * cPP + p) * cD + d];
}

// ---------------------------------------------------------------------------
// LayerNorm over D=256, one row per 256-thread block; padded rows -> 0
// ---------------------------------------------------------------------------
__global__ void __launch_bounds__(256)
k_ln(const float* __restrict__ h, const float* __restrict__ gam,
     const float* __restrict__ bet, _Float16* __restrict__ y) {
  const int row = blockIdx.x;                        // 0 .. B*PP-1
  const int rb = row % cPP;
  const int tid = threadIdx.x;
  if (rb >= cP) { y[(size_t)row * cD + tid] = (_Float16)0.f; return; }
  __shared__ float red[256];
  float v = h[(size_t)row * cD + tid];
  red[tid] = v; __syncthreads();
  for (int s = 128; s > 0; s >>= 1) { if (tid < s) red[tid] += red[tid + s]; __syncthreads(); }
  float mean = red[0] * (1.f / 256.f);
  __syncthreads();
  float d = v - mean;
  red[tid] = d * d; __syncthreads();
  for (int s = 128; s > 0; s >>= 1) { if (tid < s) red[tid] += red[tid + s]; __syncthreads(); }
  float var = red[0] * (1.f / 256.f);
  float rstd = rsqrtf(var + 1e-5f);
  y[(size_t)row * cD + tid] = (_Float16)(d * rstd * gam[tid] + bet[tid]);
}

// ---------------------------------------------------------------------------
// WMMA GEMM (NT): C[M,N] = A[M,K] * Bt[N,K]^T, f16 in / f32 accumulate.
// One wave (32 threads) computes a 32x32 tile. K % 32 == 0, M,N % 32 == 0.
// If available, the next K-tile of A and B is staged into LDS with
// GLOBAL_LOAD_ASYNC_TO_LDS_B128 (double buffered, s_wait_asynccnt) while the
// current tile feeds the WMMAs; otherwise direct loads + global_prefetch.
// Epilogues:
//  0: f16 store           outH[row*ldo+col]
//  1: f16 transposed      outH[(b*ncolsTot+col)*ppad + row%rpb]   (V -> vT)
//  2: BN+ReLU -> f16      p0..p3 = g,b,mean,var
//  3: BN+ReLU -> f32 h with +1 row offset (stem2): rows per batch = rpb
//  4: residual f32 accum  outF[row*ldo+col] += acc + (p0?p0[col]:0)
//  5: bias + exact GELU -> f16
// ---------------------------------------------------------------------------
template <int EPI>
__global__ void __launch_bounds__(32)
k_gemm(const _Float16* __restrict__ A, const _Float16* __restrict__ Bt, int K,
       float* __restrict__ outF, _Float16* __restrict__ outH, int ldo,
       const float* __restrict__ p0, const float* __restrict__ p1,
       const float* __restrict__ p2, const float* __restrict__ p3,
       int rpb, int ppad, int ncolsTot) {
  const int lane = threadIdx.x;
  const int nl = lane & 15, g = lane >> 4;
  const long m0 = (long)blockIdx.y * 32;
  const long n0 = (long)blockIdx.x * 32;

  v8f c00 = v8fzero(), c01 = v8fzero(), c10 = v8fzero(), c11 = v8fzero();

#if USE_ASYNC_LDS
  __shared__ __align__(16) _Float16 sA[2][32][32];
  __shared__ __align__(16) _Float16 sB[2][32][32];
  const int crow = lane >> 2;        // 0..7
  const int cofs = (lane & 3) * 8;   // halves (16B) within a 32-half row
  auto stage = [&](int buf, int kk) {
#pragma unroll
    for (int t = 0; t < 4; ++t) {
      int r = t * 8 + crow;
      async_b128(A + (size_t)(m0 + r) * K + kk + cofs, &sA[buf][r][cofs]);
      async_b128(Bt + (size_t)(n0 + r) * K + kk + cofs, &sB[buf][r][cofs]);
    }
  };
  stage(0, 0);
  int buf = 0;
  for (int kk = 0; kk < K; kk += 32) {
    if (kk + 32 < K) {
      stage(buf ^ 1, kk + 32);
      __builtin_amdgcn_s_wait_asynccnt(8);   // current tile's 8 copies done
    } else {
      __builtin_amdgcn_s_wait_asynccnt(0);
    }
    H16 a0u, a1u, b0u, b1u;
    a0u.h[0] = *(const h8*)&sA[buf][nl][g * 16];
    a0u.h[1] = *(const h8*)&sA[buf][nl][g * 16 + 8];
    a1u.h[0] = *(const h8*)&sA[buf][16 + nl][g * 16];
    a1u.h[1] = *(const h8*)&sA[buf][16 + nl][g * 16 + 8];
    b0u.h[0] = *(const h8*)&sB[buf][nl][g * 16];
    b0u.h[1] = *(const h8*)&sB[buf][nl][g * 16 + 8];
    b1u.h[0] = *(const h8*)&sB[buf][16 + nl][g * 16];
    b1u.h[1] = *(const h8*)&sB[buf][16 + nl][g * 16 + 8];
    c00 = wmma_f16(a0u.v, b0u.v, c00);
    c01 = wmma_f16(a0u.v, b1u.v, c01);
    c10 = wmma_f16(a1u.v, b0u.v, c10);
    c11 = wmma_f16(a1u.v, b1u.v, c11);
    buf ^= 1;
  }
#else
  const _Float16* a0p = A + (size_t)(m0 + nl) * K + g * 16;
  const _Float16* a1p = A + (size_t)(m0 + 16 + nl) * K + g * 16;
  const _Float16* b0p = Bt + (size_t)(n0 + nl) * K + g * 16;
  const _Float16* b1p = Bt + (size_t)(n0 + 16 + nl) * K + g * 16;
  for (int kk = 0; kk < K; kk += 32) {
    if (kk + 32 < K) {
      __builtin_prefetch(a0p + kk + 32, 0, 3);
      __builtin_prefetch(a1p + kk + 32, 0, 3);
      __builtin_prefetch(b0p + kk + 32, 0, 3);
      __builtin_prefetch(b1p + kk + 32, 0, 3);
    }
    H16 a0u, a1u, b0u, b1u;
    a0u.h[0] = *(const h8*)(a0p + kk);
    a0u.h[1] = *(const h8*)(a0p + kk + 8);
    a1u.h[0] = *(const h8*)(a1p + kk);
    a1u.h[1] = *(const h8*)(a1p + kk + 8);
    b0u.h[0] = *(const h8*)(b0p + kk);
    b0u.h[1] = *(const h8*)(b0p + kk + 8);
    b1u.h[0] = *(const h8*)(b1p + kk);
    b1u.h[1] = *(const h8*)(b1p + kk + 8);
    c00 = wmma_f16(a0u.v, b0u.v, c00);
    c01 = wmma_f16(a0u.v, b1u.v, c01);
    c10 = wmma_f16(a1u.v, b0u.v, c10);
    c11 = wmma_f16(a1u.v, b1u.v, c11);
  }
#endif

  v8f cc[2][2] = {{c00, c01}, {c10, c11}};
#pragma unroll
  for (int i = 0; i < 2; ++i) {
#pragma unroll
    for (int j = 0; j < 2; ++j) {
#pragma unroll
      for (int r = 0; r < 8; ++r) {
        long row = m0 + i * 16 + g * 8 + r;
        long col = n0 + j * 16 + nl;
        float acc = cc[i][j][r];
        if constexpr (EPI == 0) {
          outH[(size_t)row * ldo + col] = (_Float16)acc;
        } else if constexpr (EPI == 1) {
          long b = row / rpb, rr = row % rpb;
          outH[((size_t)b * ncolsTot + col) * ppad + rr] = (_Float16)acc;
        } else if constexpr (EPI == 2) {
          float sc = p0[col] * rsqrtf(p3[col] + 1e-5f);
          float sh = p1[col] - p2[col] * sc;
          float vv = acc * sc + sh;
          outH[(size_t)row * ldo + col] = (_Float16)(vv > 0.f ? vv : 0.f);
        } else if constexpr (EPI == 3) {
          float sc = p0[col] * rsqrtf(p3[col] + 1e-5f);
          float sh = p1[col] - p2[col] * sc;
          float vv = acc * sc + sh;
          long b = row / rpb, rr = row % rpb;
          outF[((size_t)b * ppad + 1 + rr) * ldo + col] = (vv > 0.f ? vv : 0.f);
        } else if constexpr (EPI == 4) {
          float add = p0 ? p0[col] : 0.f;
          outF[(size_t)row * ldo + col] += acc + add;
        } else if constexpr (EPI == 5) {
          float t = acc + p0[col];
          float gl = 0.5f * t * (1.f + erff(t * 0.70710678f));
          outH[(size_t)row * ldo + col] = (_Float16)gl;
        }
      }
    }
  }
}

// ---------------------------------------------------------------------------
// Attention: one wave per (b, head, 16-query-row tile). Online softmax over
// key tiles of 32; S = Q*K^T and O += P*V both via v_wmma_f32_16x16x32_f16.
// Probs transposed (C-layout -> A-layout) through LDS within the wave.
// Reproduces softmax_with_policy incl. the eps/n * sum(V) term.
// ---------------------------------------------------------------------------
__global__ void __launch_bounds__(32)
k_attn(const _Float16* __restrict__ q, const _Float16* __restrict__ k,
       const _Float16* __restrict__ vT, const float* __restrict__ policy,
       _Float16* __restrict__ o) {
  const int lane = threadIdx.x;
  const int nl = lane & 15, g = lane >> 4, mb = g * 8;
  const int mt = blockIdx.x, hh = blockIdx.y, b = blockIdx.z;
  const int m0 = mt * 16;

  __shared__ __align__(16) _Float16 probs[16][32];

  // Q fragment (A operand): DK=16 so depth 16..31 is zero-padded
  H16 qa;
  {
    const _Float16* qp = q + ((size_t)b * cPP + m0 + nl) * cHDK + hh * cDK;
    qa.h[0] = g ? h8zero() : ((const h8*)qp)[0];
    qa.h[1] = g ? h8zero() : ((const h8*)qp)[1];
  }

  v8f o0 = v8fzero(), o1 = v8fzero();
  float mrun[8], lrun[8];
#pragma unroll
  for (int r = 0; r < 8; ++r) { mrun[r] = -1e30f; lrun[r] = 0.f; }
  float vs0 = 0.f, vs1 = 0.f;

  for (int t0 = 0; t0 < cPP; t0 += 32) {
    // --- scores: two 16-key WMMAs (B depth 16..31 unused -> zero lanes) ---
    const _Float16* kbase = k + ((size_t)b * cPP + t0 + nl) * cHDK + hh * cDK;
    H16 kb0, kb1;
    if (lane < 16) {
      const _Float16* kp1 = kbase + (size_t)16 * cHDK;
      kb0.h[0] = ((const h8*)kbase)[0]; kb0.h[1] = ((const h8*)kbase)[1];
      kb1.h[0] = ((const h8*)kp1)[0];   kb1.h[1] = ((const h8*)kp1)[1];
    } else {
      kb0.h[0] = h8zero(); kb0.h[1] = h8zero();
      kb1.h[0] = h8zero(); kb1.h[1] = h8zero();
    }
    v8f s0 = wmma_f16(qa.v, kb0.v, v8fzero());
    v8f s1 = wmma_f16(qa.v, kb1.v, v8fzero());

    const int c0 = t0 + nl, c1 = t0 + 16 + nl;
    const bool vld0 = c0 < cP, vld1 = c1 < cP;
    const float ap0 = vld0 ? policy[(size_t)b * cP + c0] : 0.f;
    const float ap1 = vld1 ? policy[(size_t)b * cP + c1] : 0.f;

    float e0[8], e1[8];
#pragma unroll
    for (int r = 0; r < 8; ++r) {
      float x0 = vld0 ? s0[r] * 0.25f : -1e30f;   // scale = 1/sqrt(DK)
      float x1 = vld1 ? s1[r] * 0.25f : -1e30f;
      float mx = fmaxf(x0, x1);
      for (int off = 1; off < 16; off <<= 1) mx = fmaxf(mx, __shfl_xor(mx, off, 32));
      float mnew = fmaxf(mrun[r], mx);
      float alpha = expf(mrun[r] - mnew);
      int row = m0 + mb + r;
      float a0 = (c0 == row) ? 1.f : ap0;         // diag: policy + (1-policy)
      float a1 = (c1 == row) ? 1.f : ap1;
      float ee0 = expf(x0 - mnew) * a0;
      float ee1 = expf(x1 - mnew) * a1;
      float es = ee0 + ee1;
      for (int off = 1; off < 16; off <<= 1) es += __shfl_xor(es, off, 32);
      lrun[r] = lrun[r] * alpha + es;
      mrun[r] = mnew;
      o0[r] *= alpha;
      o1[r] *= alpha;
      e0[r] = ee0; e1[r] = ee1;
    }

    // --- C-layout -> A-layout transpose of probs through LDS ---
#pragma unroll
    for (int r = 0; r < 8; ++r) {
      probs[mb + r][nl] = (_Float16)e0[r];
      probs[mb + r][16 + nl] = (_Float16)e1[r];
    }
    __syncthreads();
    H16 pa;
    pa.h[0] = *(const h8*)(&probs[nl][g * 16]);
    pa.h[1] = *(const h8*)(&probs[nl][g * 16 + 8]);

    // --- V fragments (B operand from transposed V) + column sums ---
    const _Float16* vp0 = vT + ((size_t)b * cHDV + hh * cDV + nl) * cPP + t0 + g * 16;
    const _Float16* vp1 = vp0 + (size_t)16 * cPP;
    H16 vb0, vb1;
    vb0.h[0] = ((const h8*)vp0)[0]; vb0.h[1] = ((const h8*)vp0)[1];
    vb1.h[0] = ((const h8*)vp1)[0]; vb1.h[1] = ((const h8*)vp1)[1];
#pragma unroll
    for (int e = 0; e < 16; ++e) { vs0 += (float)vb0.v[e]; vs1 += (float)vb1.v[e]; }

    if (t0 + 32 < cPP) {             // pull next K/V tiles toward the WGP
      __builtin_prefetch(kbase + (size_t)32 * cHDK, 0, 3);
      __builtin_prefetch(vp0 + 32, 0, 3);
      __builtin_prefetch(vp1 + 32, 0, 3);
    }

    o0 = wmma_f16(pa.v, vb0.v, o0);
    o1 = wmma_f16(pa.v, vb1.v, o1);
    __syncthreads();
  }

  vs0 += __shfl_xor(vs0, 16, 32);
  vs1 += __shfl_xor(vs1, 16, 32);
  const float epsN = 1e-6f / (float)cP;
#pragma unroll
  for (int r = 0; r < 8; ++r) {
    int row = m0 + mb + r;
    if (row < cP) {
      float denom = lrun[r] + 1e-6f;
      _Float16* op = o + ((size_t)b * cPP + row) * cHDV + hh * cDV;
      op[nl] = (_Float16)((o0[r] + epsN * vs0) / denom);
      op[16 + nl] = (_Float16)((o1[r] + epsN * vs1) / denom);
    }
  }
}

// ---------------------------------------------------------------------------
// Host driver
// ---------------------------------------------------------------------------
static inline size_t alignUp256(size_t x) { return (x + 255) & ~(size_t)255; }

extern "C" void kernel_launch(void* const* d_in, const int* in_sizes, int n_in,
                              void* d_out, int out_size, void* d_ws, size_t ws_size,
                              hipStream_t stream) {
  (void)in_sizes; (void)n_in; (void)out_size; (void)ws_size;
  const float* x       = (const float*)d_in[0];
  const float* policy  = (const float*)d_in[1];
  const float* conv1_w = (const float*)d_in[2];
  const float* conv2_w = (const float*)d_in[3];
  const float* bn1_g = (const float*)d_in[4];
  const float* bn1_b = (const float*)d_in[5];
  const float* bn1_m = (const float*)d_in[6];
  const float* bn1_v = (const float*)d_in[7];
  const float* bn2_g = (const float*)d_in[8];
  const float* bn2_b = (const float*)d_in[9];
  const float* bn2_m = (const float*)d_in[10];
  const float* bn2_v = (const float*)d_in[11];
  const float* cls   = (const float*)d_in[12];
  const float* ln1_g = (const float*)d_in[13];
  const float* ln1_b = (const float*)d_in[14];
  const float* ln2_g = (const float*)d_in[15];
  const float* ln2_b = (const float*)d_in[16];
  const float* wq    = (const float*)d_in[17];
  const float* wk    = (const float*)d_in[18];
  const float* wv    = (const float*)d_in[19];
  const float* fc_w  = (const float*)d_in[20];
  const float* ff_w1 = (const float*)d_in[21];
  const float* ff_b1 = (const float*)d_in[22];
  const float* ff_w2 = (const float*)d_in[23];
  const float* ff_b2 = (const float*)d_in[24];
  float* out = (float*)d_out;

  char* ws = (char*)d_ws;
  size_t off = 0;
  auto alloc = [&](size_t bytes) -> void* {
    void* p = ws + off;
    off = alignUp256(off + bytes);
    return p;
  };

  _Float16* c1w16  = (_Float16*)alloc((size_t)cD * cC * 2);
  _Float16* c2w16  = (_Float16*)alloc((size_t)cD * cD * 2);
  _Float16* wq16   = (_Float16*)alloc((size_t)cNB * cHDK * cD * 2);
  _Float16* wk16   = (_Float16*)alloc((size_t)cNB * cHDK * cD * 2);
  _Float16* wv16   = (_Float16*)alloc((size_t)cNB * cHDV * cD * 2);
  _Float16* fc16   = (_Float16*)alloc((size_t)cNB * cD * cHDV * 2);
  _Float16* ffw116 = (_Float16*)alloc((size_t)cNB * 2 * cD * cD * 2);
  _Float16* ffw216 = (_Float16*)alloc((size_t)cNB * cD * 2 * cD * 2);
  _Float16* xt16   = (_Float16*)alloc((size_t)cB * cN * cC * 2);
  _Float16* h1t16  = (_Float16*)alloc((size_t)cB * cN * cD * 2);
  float*    hbuf   = (float*)   alloc((size_t)cB * cPP * cD * 4);
  _Float16* y16    = (_Float16*)alloc((size_t)cB * cPP * cD * 2);
  _Float16* q16    = (_Float16*)alloc((size_t)cB * cPP * cHDK * 2);
  _Float16* k16    = (_Float16*)alloc((size_t)cB * cPP * cHDK * 2);
  _Float16* vT16   = (_Float16*)alloc((size_t)cB * cHDV * cPP * 2);
  _Float16* o16    = (_Float16*)alloc((size_t)cB * cPP * cHDV * 2);
  _Float16* z16    = (_Float16*)alloc((size_t)cB * cPP * 2 * cD * 2);

  auto cvt = [&](const float* s, _Float16* d, int n) {
    k_cvt<<<(n + 255) / 256, 256, 0, stream>>>(s, d, n);
  };
  cvt(conv1_w, c1w16, cD * cC);
  cvt(conv2_w, c2w16, cD * cD);
  cvt(wq, wq16, cNB * cHDK * cD);
  cvt(wk, wk16, cNB * cHDK * cD);
  cvt(wv, wv16, cNB * cHDV * cD);
  cvt(fc_w, fc16, cNB * cD * cHDV);
  cvt(ff_w1, ffw116, cNB * 2 * cD * cD);
  cvt(ff_w2, ffw216, cNB * cD * 2 * cD);

  // stem
  k_transpose_x<<<(cB * cN * cC + 255) / 256, 256, 0, stream>>>(x, xt16);
  const int Ms = cB * cN;       // 8192
  const int Mt = cB * cPP;      // 8448
  dim3 blk(32, 1, 1);
  // conv1 + bn1 + relu -> h1t f16
  k_gemm<2><<<dim3(cD / 32, Ms / 32), blk, 0, stream>>>(
      xt16, c1w16, cC, nullptr, h1t16, cD, bn1_g, bn1_b, bn1_m, bn1_v, 0, 0, 0);
  // conv2 + bn2 + relu -> h f32 (rows shifted +1 for cls)
  k_gemm<3><<<dim3(cD / 32, Ms / 32), blk, 0, stream>>>(
      h1t16, c2w16, cD, hbuf, nullptr, cD, bn2_g, bn2_b, bn2_m, bn2_v, cN, cPP, 0);
  k_cls<<<(cB * cD + 255) / 256, 256, 0, stream>>>(cls, hbuf);

  for (int i = 0; i < cNB; ++i) {
    // LN1
    k_ln<<<cB * cPP, 256, 0, stream>>>(hbuf, ln1_g + i * cD, ln1_b + i * cD, y16);
    // Q, K projections
    k_gemm<0><<<dim3(cHDK / 32, Mt / 32), blk, 0, stream>>>(
        y16, wq16 + (size_t)i * cHDK * cD, cD, nullptr, q16, cHDK,
        nullptr, nullptr, nullptr, nullptr, 0, 0, 0);
    k_gemm<0><<<dim3(cHDK / 32, Mt / 32), blk, 0, stream>>>(
        y16, wk16 + (size_t)i * cHDK * cD, cD, nullptr, k16, cHDK,
        nullptr, nullptr, nullptr, nullptr, 0, 0, 0);
    // V projection (stored transposed [feature][token])
    k_gemm<1><<<dim3(cHDV / 32, Mt / 32), blk, 0, stream>>>(
        y16, wv16 + (size_t)i * cHDV * cD, cD, nullptr, vT16, 0,
        nullptr, nullptr, nullptr, nullptr, cPP, cPP, cHDV);
    // attention
    k_attn<<<dim3(cPP / 16, cH, cB), blk, 0, stream>>>(q16, k16, vT16, policy, o16);
    // output projection, residual into h
    k_gemm<4><<<dim3(cD / 32, Mt / 32), blk, 0, stream>>>(
        o16, fc16 + (size_t)i * cD * cHDV, cHDV, hbuf, nullptr, cD,
        nullptr, nullptr, nullptr, nullptr, 0, 0, 0);
    // LN2
    k_ln<<<cB * cPP, 256, 0, stream>>>(hbuf, ln2_g + i * cD, ln2_b + i * cD, y16);
    // FF1 + bias + gelu -> z16
    k_gemm<5><<<dim3(2 * cD / 32, Mt / 32), blk, 0, stream>>>(
        y16, ffw116 + (size_t)i * 2 * cD * cD, cD, nullptr, z16, 2 * cD,
        ff_b1 + (size_t)i * 2 * cD, nullptr, nullptr, nullptr, 0, 0, 0);
    // FF2 + bias, residual into h
    k_gemm<4><<<dim3(cD / 32, Mt / 32), blk, 0, stream>>>(
        z16, ffw216 + (size_t)i * cD * 2 * cD, 2 * cD, hbuf, nullptr, cD,
        ff_b2 + (size_t)i * cD, nullptr, nullptr, nullptr, 0, 0, 0);
  }

  k_out<<<(int)(((size_t)cB * cP * cD + 255) / 256), 256, 0, stream>>>(hbuf, out);
}
